// DGCNN_2894807957806
// MI455X (gfx1250) — matrix operand order repaired and checked by reading, not compiled, not run on hardware
//
#include <hip/hip_runtime.h>

typedef __bf16 v16bf __attribute__((ext_vector_type(16)));
typedef float  v8f   __attribute__((ext_vector_type(8)));

#define NPTS 2048
#define KNN  20
#define NEGINF (-3.4e38f)

static __device__ __forceinline__ float leaky02(float y) {
    return y >= 0.0f ? y : 0.2f * y;
}

static __device__ __forceinline__ float fmaxf2(float a, float b) {
    return a > b ? a : b;
}

// WMMA fragment index helpers (wave32; A/B use contiguous k = e + 16h, not the ISA table order):
//  A (16x32 bf16): lane half h, row = lane%16, elem e -> K
//  B (32x16 bf16): col = lane%16, elem e -> K = e + 16h
//  C/D (16x16 f32): col = lane%16, row = vgpr + 8h
static __device__ __forceinline__ int a_k_of_elem(int e, int h) {
    return e + 16 * h;
}

// ---------------------------------------------------------------------------
// Squared norms per point: sq[b*N+n] = sum_c in[c*N+n]^2
// ---------------------------------------------------------------------------
template <int C>
__global__ void sqnorm_kernel(const float* __restrict__ in, long batchStride,
                              float* __restrict__ sq) {
    int n = blockIdx.x * blockDim.x + threadIdx.x;
    int b = blockIdx.y;
    if (n >= NPTS) return;
    const float* p = in + (long)b * batchStride;
    float acc = 0.0f;
#pragma unroll
    for (int c = 0; c < C; ++c) {
        float v = p[(long)c * NPTS + n];
        acc += v * v;
    }
    sq[(long)b * NPTS + n] = acc;
}

// ---------------------------------------------------------------------------
// Pairwise "distance" (2*x^T x - |x|^2 - |x|^2) for ONE batch via bf16 WMMA.
// One wave computes a 16x64 tile: shared A fragment, 4 B fragments,
// 4 accumulators -> 4 v_wmma per K chunk.
// ---------------------------------------------------------------------------
template <int C>
__global__ __launch_bounds__(32) void dist_wmma_kernel(
    const float* __restrict__ xin, const float* __restrict__ sq,
    float* __restrict__ dist) {
    const int n0 = blockIdx.x * 16;
    const int m0 = blockIdx.y * 64;
    const int l  = threadIdx.x & 31;
    const int h  = l >> 4;
    const int lm = l & 15;

    v8f acc[4] = {v8f{}, v8f{}, v8f{}, v8f{}};
    for (int kc = 0; kc < C; kc += 32) {
        v16bf a;
#pragma unroll
        for (int e = 0; e < 16; ++e) {
            int cc = kc + a_k_of_elem(e, h);
            float av = 0.0f;
            if ((C & 31) == 0 || cc < C)           // folds away for C=64/128
                av = xin[(long)cc * NPTS + (n0 + lm)];
            a[e] = (__bf16)av;
        }
#pragma unroll
        for (int j = 0; j < 4; ++j) {
            v16bf bf;
#pragma unroll
            for (int e = 0; e < 16; ++e) {
                int cc = kc + e + 16 * h;
                float bv = 0.0f;
                if ((C & 31) == 0 || cc < C)
                    bv = xin[(long)cc * NPTS + (m0 + 16 * j + lm)];
                bf[e] = (__bf16)bv;
            }
            acc[j] = __builtin_amdgcn_wmma_f32_16x16x32_bf16(
                false, a, false, bf, (short)0, acc[j], false, false);
        }
    }

#pragma unroll
    for (int j = 0; j < 4; ++j) {
        const int   m   = m0 + 16 * j + lm;
        const float sqm = sq[m];
#pragma unroll
        for (int v = 0; v < 8; ++v) {
            int n = n0 + v + 8 * h;
            dist[(long)n * NPTS + m] = 2.0f * acc[j][v] - sq[n] - sqm;
        }
    }
}

// ---------------------------------------------------------------------------
// Top-k (k=20) per row: LDS-resident row + 20 block-argmax sweeps.
// ---------------------------------------------------------------------------
__global__ __launch_bounds__(256) void topk_kernel(
    const float* __restrict__ dist, int* __restrict__ idx) {
    __shared__ float row[NPTS];
    __shared__ float rv[256];
    __shared__ int   ri[256];
    const int n = blockIdx.x;
    const int t = threadIdx.x;

    for (int j = t; j < NPTS; j += 256) row[j] = dist[(long)n * NPTS + j];
    __syncthreads();

    for (int kk = 0; kk < KNN; ++kk) {
        float bv = NEGINF;
        int   bi = 0;
        for (int j = t; j < NPTS; j += 256) {
            float v = row[j];
            if (v > bv) { bv = v; bi = j; }
        }
        rv[t] = bv; ri[t] = bi;
        __syncthreads();
        for (int s2 = 128; s2 > 0; s2 >>= 1) {
            if (t < s2 && rv[t + s2] > rv[t]) { rv[t] = rv[t + s2]; ri[t] = ri[t + s2]; }
            __syncthreads();
        }
        if (t == 0) {
            idx[(long)n * KNN + kk] = ri[0];
            row[ri[0]] = NEGINF;
        }
        __syncthreads();
    }
}

// ---------------------------------------------------------------------------
// EdgeConv: one wave per (point n, 16 output channels, batch b).
// D rows = neighbor slot (k=20 padded to 32 -> two 16-row tiles),
// D cols = output channel, K = 2C edge-feature channels ([feat-ctr, ctr]).
// Gather is branch-free: padded slots clamp to neighbor 0 and are excluded
// in the max epilogue; feature-vs-center picked with selects, not branches.
// ---------------------------------------------------------------------------
template <int C>
__global__ __launch_bounds__(32) void edgeconv_wmma_kernel(
    const float* __restrict__ hin, long inBatchStride,
    const float* __restrict__ W, const float* __restrict__ s,
    const float* __restrict__ t, const int* __restrict__ idx,
    float* __restrict__ out) {
    constexpr int C2 = 2 * C;
    const int n  = blockIdx.x;
    const int o0 = blockIdx.y * 16;
    const int b  = blockIdx.z;
    const int l  = threadIdx.x & 31;
    const int h  = l >> 4;
    const int lm = l & 15;

    const float* hb  = hin + (long)b * inBatchStride;
    const int*   nbr = idx + ((long)b * NPTS + n) * KNN;
    const int mA = nbr[lm];                                   // slot lm < 20 always
    const int mB = nbr[lm < (KNN - 16) ? 16 + lm : 0];        // clamped; masked in epilogue

    v8f acc0 = {}, acc1 = {};
    for (int kc = 0; kc < C2; kc += 32) {
        v16bf bw;
#pragma unroll
        for (int e = 0; e < 16; ++e) {
            int cc = kc + e + 16 * h;
            float wv = 0.0f;
            if ((C2 & 31) == 0 || cc < C2)         // folds away for C2=128/256
                wv = W[(long)(o0 + lm) * C2 + cc];
            bw[e] = (__bf16)wv;
        }
        v16bf a0, a1;
#pragma unroll
        for (int e = 0; e < 16; ++e) {
            int cc = kc + a_k_of_elem(e, h);
            float v0 = 0.0f, v1 = 0.0f;
            if ((C2 & 31) == 0 || cc < C2) {
                int  ce     = (cc < C) ? cc : (cc - C);
                bool isdiff = (cc < C);
                float ctr = hb[(long)ce * NPTS + n];
                float fA  = hb[(long)ce * NPTS + mA];
                float fB  = hb[(long)ce * NPTS + mB];
                v0 = isdiff ? (fA - ctr) : ctr;
                v1 = isdiff ? (fB - ctr) : ctr;
            }
            a0[e] = (__bf16)v0;
            a1[e] = (__bf16)v1;
        }
        acc0 = __builtin_amdgcn_wmma_f32_16x16x32_bf16(
            false, a0, false, bw, (short)0, acc0, false, false);
        acc1 = __builtin_amdgcn_wmma_f32_16x16x32_bf16(
            false, a1, false, bw, (short)0, acc1, false, false);
    }

    const int   o  = o0 + lm;
    const float sv = s[o];
    const float tv = t[o];
    float mx = NEGINF;
#pragma unroll
    for (int v = 0; v < 8; ++v) {
        int r0 = v + 8 * h;        // neighbor slot, tile 0 (always < 16 < KNN)
        int r1 = 16 + v + 8 * h;   // neighbor slot, tile 1
        float y0 = leaky02(acc0[v] * sv + tv);
        float y1 = leaky02(acc1[v] * sv + tv);
        (void)r0;
        mx = fmaxf2(mx, y0);
        if (r1 < KNN) mx = fmaxf2(mx, y1);
    }
    float other = __shfl_xor(mx, 16, 32);
    mx = fmaxf2(mx, other);
    if (h == 0)
        out[((long)b * 512 + o) * (long)NPTS + n] = mx;   // out pre-offset by co*N
}

// ---------------------------------------------------------------------------
// Pointwise conv 512->1024 + BN + LeakyReLU via bf16 WMMA.
// One wave computes 16 channels x 64 points: shared W fragment, 4 B tiles.
// ---------------------------------------------------------------------------
__global__ __launch_bounds__(32) void conv1d_wmma_kernel(
    const float* __restrict__ FB, const float* __restrict__ W,
    const float* __restrict__ s, const float* __restrict__ t,
    float* __restrict__ Z) {
    const int CIN = 512;
    const int n0 = blockIdx.x * 64;
    const int o0 = blockIdx.y * 16;
    const int b  = blockIdx.z;
    const int l  = threadIdx.x & 31;
    const int h  = l >> 4;
    const int lm = l & 15;

    const float* fb = FB + (long)b * CIN * NPTS;
    v8f acc[4] = {v8f{}, v8f{}, v8f{}, v8f{}};
    for (int kc = 0; kc < CIN; kc += 32) {
        if (kc + 32 < CIN)
            __builtin_prefetch(&fb[(long)(kc + 32) * NPTS + n0 + lm], 0, 1);
        v16bf aw;
#pragma unroll
        for (int e = 0; e < 16; ++e) {
            int k = a_k_of_elem(e, h);
            aw[e] = (__bf16)W[(long)(o0 + lm) * CIN + (kc + k)];
        }
#pragma unroll
        for (int j = 0; j < 4; ++j) {
            v16bf bx;
#pragma unroll
            for (int e = 0; e < 16; ++e) {
                bx[e] = (__bf16)fb[(long)(kc + e + 16 * h) * NPTS + (n0 + 16 * j + lm)];
            }
            acc[j] = __builtin_amdgcn_wmma_f32_16x16x32_bf16(
                false, aw, false, bx, (short)0, acc[j], false, false);
        }
    }
#pragma unroll
    for (int j = 0; j < 4; ++j) {
#pragma unroll
        for (int v = 0; v < 8; ++v) {
            int o = o0 + v + 8 * h;
            float y = leaky02(acc[j][v] * s[o] + t[o]);
            Z[((long)b * 1024 + o) * (long)NPTS + (n0 + 16 * j + lm)] = y;
        }
    }
}

// ---------------------------------------------------------------------------
// Global max + mean pool over N per (b, channel).
// ---------------------------------------------------------------------------
__global__ __launch_bounds__(256) void pool_kernel(
    const float* __restrict__ Z, float* __restrict__ P) {
    const int o = blockIdx.x;
    const int b = blockIdx.y;
    const int t = threadIdx.x;
    const float* z = Z + ((long)b * 1024 + o) * (long)NPTS;
    __shared__ float smx[256], ssm[256];
    float mx = NEGINF, sm = 0.0f;
    for (int j = t; j < NPTS; j += 256) {
        float v = z[j];
        mx = fmaxf2(mx, v);
        sm += v;
    }
    smx[t] = mx; ssm[t] = sm;
    __syncthreads();
    for (int s2 = 128; s2 > 0; s2 >>= 1) {
        if (t < s2) {
            smx[t] = fmaxf2(smx[t], smx[t + s2]);
            ssm[t] += ssm[t + s2];
        }
        __syncthreads();
    }
    if (t == 0) {
        P[(long)b * 2048 + o]        = smx[0];
        P[(long)b * 2048 + 1024 + o] = ssm[0] * (1.0f / (float)NPTS);
    }
}

// ---------------------------------------------------------------------------
// Generic small FC: out[b,o] = act((in[b,:]·W[o,:] + bias?)*s + t)
// ---------------------------------------------------------------------------
__global__ __launch_bounds__(128) void fc_kernel(
    const float* __restrict__ in, int KDIM,
    const float* __restrict__ W, const float* __restrict__ bias,
    const float* __restrict__ s, const float* __restrict__ t,
    int leaky, float* __restrict__ out, int ODIM) {
    const int o  = blockIdx.x;
    const int b  = blockIdx.y;
    const int ti = threadIdx.x;
    const float* iv = in + (long)b * KDIM;
    const float* wv = W + (long)o * KDIM;
    float acc = 0.0f;
    for (int j = ti; j < KDIM; j += 128) acc += iv[j] * wv[j];
    __shared__ float red[128];
    red[ti] = acc;
    __syncthreads();
    for (int s2 = 64; s2 > 0; s2 >>= 1) {
        if (ti < s2) red[ti] += red[ti + s2];
        __syncthreads();
    }
    if (ti == 0) {
        float y = red[0];
        if (bias) y += bias[o];
        if (s)    y = y * s[o] + t[o];
        if (leaky) y = leaky02(y);
        out[(long)b * ODIM + o] = y;
    }
}

// ---------------------------------------------------------------------------
// Host-side per-layer launcher (compile-time C, O)
// ---------------------------------------------------------------------------
template <int C, int O>
static inline void launch_layer(const float* inp, long instride,
                                const float* W, const float* s, const float* t,
                                float* SQ, float* DIST, int* IDX, float* out,
                                hipStream_t stream) {
    const int B = 8;
    sqnorm_kernel<C><<<dim3(NPTS / 256, B), 256, 0, stream>>>(inp, instride, SQ);
    for (int b = 0; b < B; ++b) {
        dist_wmma_kernel<C><<<dim3(NPTS / 16, NPTS / 64), 32, 0, stream>>>(
            inp + (long)b * instride, SQ + (long)b * NPTS, DIST);
        topk_kernel<<<NPTS, 256, 0, stream>>>(DIST, IDX + (long)b * NPTS * KNN);
    }
    edgeconv_wmma_kernel<C><<<dim3(NPTS, O / 16, B), 32, 0, stream>>>(
        inp, instride, W, s, t, IDX, out);
}

// ---------------------------------------------------------------------------
extern "C" void kernel_launch(void* const* d_in, const int* in_sizes, int n_in,
                              void* d_out, int out_size, void* d_ws, size_t ws_size,
                              hipStream_t stream) {
    (void)in_sizes; (void)n_in; (void)out_size; (void)ws_size;
    const int B = 8;

    const float* x     = (const float*)d_in[0];
    const float* w_c1d = (const float*)d_in[13];
    const float* s4    = (const float*)d_in[14];
    const float* t4    = (const float*)d_in[15];
    const float* w_l1  = (const float*)d_in[16];
    const float* s5    = (const float*)d_in[17];
    const float* t5    = (const float*)d_in[18];
    const float* w_l2  = (const float*)d_in[19];
    const float* b_l2  = (const float*)d_in[20];
    const float* s6    = (const float*)d_in[21];
    const float* t6    = (const float*)d_in[22];
    const float* w_l3  = (const float*)d_in[23];
    const float* b_l3  = (const float*)d_in[24];

    // Workspace layout (float offsets)
    float* ws   = (float*)d_ws;
    float* FB   = ws;                    // (B,512,N)   concat of layer outputs, 32 MB
    float* Z    = ws + 8388608;          // (B,1024,N)  conv1d out, 64 MB
    float* DIST = ws + 25165824;         // (N,N)       per-batch distance, 16 MB
    float* SQ   = ws + 29360128;         // (B,N)
    float* POOL = ws + 29376512;         // (B,2048)
    float* F1   = ws + 29392896;         // (B,512)
    float* F2   = ws + 29396992;         // (B,256)
    int*   IDX  = (int*)(ws + 29399040); // (B,N,K)

    // Layer 0: C=3  -> O=64  (out at FB channel 0)
    launch_layer<3, 64>(x, 3L * NPTS,
                        (const float*)d_in[1], (const float*)d_in[2], (const float*)d_in[3],
                        SQ, DIST, IDX, FB + 0L * NPTS, stream);
    // Layer 1: C=64 -> O=64  (in FB ch0, out FB ch64)
    launch_layer<64, 64>(FB + 0L * NPTS, 512L * NPTS,
                         (const float*)d_in[4], (const float*)d_in[5], (const float*)d_in[6],
                         SQ, DIST, IDX, FB + 64L * NPTS, stream);
    // Layer 2: C=64 -> O=128 (in FB ch64, out FB ch128)
    launch_layer<64, 128>(FB + 64L * NPTS, 512L * NPTS,
                          (const float*)d_in[7], (const float*)d_in[8], (const float*)d_in[9],
                          SQ, DIST, IDX, FB + 128L * NPTS, stream);
    // Layer 3: C=128 -> O=256 (in FB ch128, out FB ch256)
    launch_layer<128, 256>(FB + 128L * NPTS, 512L * NPTS,
                           (const float*)d_in[10], (const float*)d_in[11], (const float*)d_in[12],
                           SQ, DIST, IDX, FB + 256L * NPTS, stream);

    conv1d_wmma_kernel<<<dim3(NPTS / 64, 64, B), 32, 0, stream>>>(FB, w_c1d, s4, t4, Z);
    pool_kernel<<<dim3(1024, B), 256, 0, stream>>>(Z, POOL);
    fc_kernel<<<dim3(512, B), 128, 0, stream>>>(POOL, 2048, w_l1, nullptr, s5, t5, 1, F1, 512);
    fc_kernel<<<dim3(256, B), 128, 0, stream>>>(F1, 512, w_l2, b_l2, s6, t6, 1, F2, 256);
    fc_kernel<<<dim3(40, B), 128, 0, stream>>>(F2, 256, w_l3, b_l3, nullptr, nullptr, 0,
                                               (float*)d_out, 40);
}
